// HeteroGNN_8478265442841
// MI455X (gfx1250) — matrix operand order repaired and checked by reading, not compiled, not run on hardware
//
#include <hip/hip_runtime.h>

// HeteroGNN (2-layer R-SAGE) for MI455X / gfx1250.
// Stages per layer:
//   1) memset aggregate buffers
//   2) pack weights (f32 -> bf16 B-fragment layout, summing Wr's for the dst GEMM)
//   3) per-relation scatter: agg[dst] += x[src] (global_atomic_add_f32, no-return), cnt[dst]+=1
//   4) fused multi-source WMMA GEMM: out = sum_s diag(1/cnt_s) * (A_s @ B_s) + bias, relu
// Requires ws_size >= ~1.25 GB.

typedef __attribute__((ext_vector_type(16))) __bf16 v16bf;
typedef __attribute__((ext_vector_type(8)))  float  v8f;

#define DD 256  // feature dim (fixed by the reference)

// -------------------------------------------------------------------------
// Scatter: one thread handles 4 channels of one edge. 64 threads per edge.
// -------------------------------------------------------------------------
__global__ __launch_bounds__(256) void scatter_add_kernel(
    const float* __restrict__ xsrc, const int* __restrict__ src,
    const int* __restrict__ dst, float* __restrict__ agg,
    float* __restrict__ cnt, int E)
{
    int t = blockIdx.x * 256 + threadIdx.x;
    int e = t >> 6;
    if (e >= E) return;
    int c = (t & 63) << 2;
    int s = src[e];
    int d = dst[e];
    const float4 v = *(const float4*)(xsrc + (size_t)s * DD + c);
    float* base = agg + (size_t)d * DD + c;
    atomicAdd(base + 0, v.x);
    atomicAdd(base + 1, v.y);
    atomicAdd(base + 2, v.z);
    atomicAdd(base + 3, v.w);
    if (c == 0) atomicAdd(cnt + d, 1.0f);
}

// -------------------------------------------------------------------------
// Pack a 256x256 f32 weight (optionally summing up to 3) into the bf16
// B-fragment layout for v_wmma_f32_16x16x32_bf16:
//   dst[tn*4096 + kb*512 + lane*16 + j] = W[(kb*32 + (lane>>4)*8 + (j&7) + (j>>3)*16)*256 + tn*16 + (lane&15)]
// -------------------------------------------------------------------------
__global__ __launch_bounds__(256) void pack_b_kernel(
    __bf16* __restrict__ dst, const float* __restrict__ Wa,
    const float* __restrict__ Wb, const float* __restrict__ Wc)
{
    int idx = blockIdx.x * 256 + threadIdx.x;     // < 65536
    int tn  = idx >> 12;
    int rem = idx & 4095;
    int kb  = rem >> 9;
    int l   = (rem >> 4) & 31;
    int j   = rem & 15;
    int k   = kb * 32 + (l >> 4) * 8 + (j & 7) + (j >> 3) * 16;
    int col = tn * 16 + (l & 15);
    float v = Wa[(size_t)k * DD + col];
    if (Wb) v += Wb[(size_t)k * DD + col];
    if (Wc) v += Wc[(size_t)k * DD + col];
    dst[idx] = (__bf16)v;
}

__global__ void bias_sum_kernel(float* __restrict__ out, const float* __restrict__ a,
                                const float* __restrict__ b, const float* __restrict__ c)
{
    int i = threadIdx.x;
    out[i] = a[i] + b[i] + c[i];
}

// -------------------------------------------------------------------------
// Fused multi-source GEMM. One wave computes one 16x16 output tile:
//   C = relu( sum_s scale_s(row) * (A_s[M,256] @ B_s[256,256]) + bias )
// scale_s(row) = 1/max(cnt_s[row],1) for aggregate sources, 1 for root.
// A is f32 row-major (converted to bf16 in-register); B is pre-packed bf16.
// M must be a multiple of 16 (true here: 200000 = 16*12500).
// -------------------------------------------------------------------------
__global__ __launch_bounds__(256) void gemm_wmma_kernel(
    const float* A0, const float* c0, const __bf16* B0,
    const float* A1, const float* c1, const __bf16* B1,
    const float* A2, const float* c2, const __bf16* B2,
    const float* A3, const float* c3, const __bf16* B3,
    const float* __restrict__ bias, float* __restrict__ C, int M, int relu)
{
    const int lane = threadIdx.x & 31;
    const int wave = threadIdx.x >> 5;
    const int tile = blockIdx.x * 8 + wave;
    const int tilesM = M >> 4;
    const int tm = tile >> 4;           // 16 N-tiles since D = 256
    const int tn = tile & 15;
    if (tm >= tilesM) return;

    const int mrow = tm * 16 + (lane & 15);
    const int hi   = lane >> 4;

    const float*  As[4] = {A0, A1, A2, A3};
    const float*  cs[4] = {c0, c1, c2, c3};
    const __bf16* Bs[4] = {B0, B1, B2, B3};

    v8f acc = {0.f, 0.f, 0.f, 0.f, 0.f, 0.f, 0.f, 0.f};

#pragma unroll
    for (int sidx = 0; sidx < 4; ++sidx) {
        if (As[sidx] == nullptr) continue;
        const float*  arow  = As[sidx] + (size_t)mrow * DD;
        const __bf16* bbase = Bs[sidx] + ((size_t)tn * 8 * 32 + lane) * 16;
        v8f part = {0.f, 0.f, 0.f, 0.f, 0.f, 0.f, 0.f, 0.f};
#pragma unroll
        for (int kb = 0; kb < 8; ++kb) {
            const int kbase = kb * 32 + hi * 8;
            // A fragment: elems 0-7 <- K = kbase..kbase+7, elems 8-15 <- K = kbase+16..kbase+23
            float4 a0 = *(const float4*)(arow + kbase);
            float4 a1 = *(const float4*)(arow + kbase + 4);
            float4 a2 = *(const float4*)(arow + kbase + 16);
            float4 a3 = *(const float4*)(arow + kbase + 20);
            v16bf af;
            af[0]  = (__bf16)a0.x; af[1]  = (__bf16)a0.y; af[2]  = (__bf16)a0.z; af[3]  = (__bf16)a0.w;
            af[4]  = (__bf16)a1.x; af[5]  = (__bf16)a1.y; af[6]  = (__bf16)a1.z; af[7]  = (__bf16)a1.w;
            af[8]  = (__bf16)a2.x; af[9]  = (__bf16)a2.y; af[10] = (__bf16)a2.z; af[11] = (__bf16)a2.w;
            af[12] = (__bf16)a3.x; af[13] = (__bf16)a3.y; af[14] = (__bf16)a3.z; af[15] = (__bf16)a3.w;
            v16bf bfv = *(const v16bf*)(bbase + (size_t)kb * (32 * 16));
            part = __builtin_amdgcn_wmma_f32_16x16x32_bf16(
                false, af, false, bfv, (short)0, part, false, false);
        }
        if (cs[sidx]) {
            const float* cp = cs[sidx] + tm * 16 + hi * 8;
#pragma unroll
            for (int i = 0; i < 8; ++i) acc[i] += part[i] * (1.0f / fmaxf(cp[i], 1.0f));
        } else {
#pragma unroll
            for (int i = 0; i < 8; ++i) acc[i] += part[i];
        }
    }

    const float b = bias[tn * 16 + (lane & 15)];
    float* crow = C + (size_t)(tm * 16 + hi * 8) * DD + tn * 16 + (lane & 15);
#pragma unroll
    for (int i = 0; i < 8; ++i) {
        float v = acc[i] + b;
        if (relu) v = fmaxf(v, 0.f);
        crow[(size_t)i * DD] = v;
    }
}

// -------------------------------------------------------------------------
extern "C" void kernel_launch(void* const* d_in, const int* in_sizes, int n_in,
                              void* d_out, int out_size, void* d_ws, size_t ws_size,
                              hipStream_t stream)
{
    const float* x_chem = (const float*)d_in[0];
    const float* x_dis  = (const float*)d_in[1];
    const float* Wl     = (const float*)d_in[2];   // (L,R,D,D)
    const float* bl     = (const float*)d_in[3];   // (L,R,D)
    const float* Wr     = (const float*)d_in[4];   // (L,R,D,D)
    const int* ei_cause  = (const int*)d_in[5];    // (2,E)
    const int* ei_relate = (const int*)d_in[6];
    const int* ei_rev    = (const int*)d_in[7];
    const int* ei_child  = (const int*)d_in[8];

    const int Nc = in_sizes[0] / DD;
    const int Nd = in_sizes[1] / DD;
    const int E  = in_sizes[5] / 2;
    const int R  = 4;
    const int L  = 2;

    // ---- workspace carve (256B-aligned chunks) ----
    char* w = (char*)d_ws;
    auto carve = [&](size_t bytes) -> char* {
        char* p = w;
        w += (bytes + 255) & ~(size_t)255;
        return p;
    };
    float* agg_cause  = (float*)carve((size_t)Nd * DD * 4);
    float* agg_relate = (float*)carve((size_t)Nd * DD * 4);
    float* agg_child  = (float*)carve((size_t)Nd * DD * 4);
    float* agg_rev    = (float*)carve((size_t)Nc * DD * 4);
    float* cnt_cause  = (float*)carve((size_t)Nd * 4);
    float* cnt_relate = (float*)carve((size_t)Nd * 4);
    float* cnt_child  = (float*)carve((size_t)Nd * 4);
    float* cnt_rev    = (float*)carve((size_t)Nc * 4);
    float* h_chem     = (float*)carve((size_t)Nc * DD * 4);
    float* h_dis      = (float*)carve((size_t)Nd * DD * 4);
    float* bsum_dis   = (float*)carve(DD * 4);
    __bf16* pk[6];
    for (int i = 0; i < 6; ++i) pk[i] = (__bf16*)carve((size_t)DD * DD * 2);

    const int pgrid = (DD * DD) / 256;            // 256 blocks
    const int sgrid = (E * 64 + 255) / 256;       // scatter grid

    for (int l = 0; l < L; ++l) {
        const float* xc = (l == 0) ? x_chem : h_chem;
        const float* xd = (l == 0) ? x_dis  : h_dis;
        float* oc = (l == L - 1) ? (float*)d_out : h_chem;
        float* od = (l == L - 1) ? (float*)d_out + (size_t)Nc * DD : h_dis;
        const int relu = (l < L - 1) ? 1 : 0;
        const float* Wl_l = Wl + (size_t)l * R * DD * DD;
        const float* Wr_l = Wr + (size_t)l * R * DD * DD;
        const float* bl_l = bl + (size_t)l * R * DD;

        // 1) zero aggregates
        (void)hipMemsetAsync(agg_cause,  0, (size_t)Nd * DD * 4, stream);
        (void)hipMemsetAsync(agg_relate, 0, (size_t)Nd * DD * 4, stream);
        (void)hipMemsetAsync(agg_child,  0, (size_t)Nd * DD * 4, stream);
        (void)hipMemsetAsync(agg_rev,    0, (size_t)Nc * DD * 4, stream);
        (void)hipMemsetAsync(cnt_cause,  0, (size_t)Nd * 4, stream);
        (void)hipMemsetAsync(cnt_relate, 0, (size_t)Nd * 4, stream);
        (void)hipMemsetAsync(cnt_child,  0, (size_t)Nd * 4, stream);
        (void)hipMemsetAsync(cnt_rev,    0, (size_t)Nc * 4, stream);

        // 2) pack weights (relation order: 0=cause,1=relate,2=rev,3=child)
        pack_b_kernel<<<pgrid, 256, 0, stream>>>(pk[0], Wl_l + 0 * DD * DD, nullptr, nullptr);
        pack_b_kernel<<<pgrid, 256, 0, stream>>>(pk[1], Wl_l + 1 * DD * DD, nullptr, nullptr);
        pack_b_kernel<<<pgrid, 256, 0, stream>>>(pk[2], Wl_l + 3 * DD * DD, nullptr, nullptr);
        pack_b_kernel<<<pgrid, 256, 0, stream>>>(pk[3], Wl_l + 2 * DD * DD, nullptr, nullptr);
        pack_b_kernel<<<pgrid, 256, 0, stream>>>(pk[4], Wr_l + 0 * DD * DD,
                                                 Wr_l + 1 * DD * DD, Wr_l + 3 * DD * DD);
        pack_b_kernel<<<pgrid, 256, 0, stream>>>(pk[5], Wr_l + 2 * DD * DD, nullptr, nullptr);
        bias_sum_kernel<<<1, DD, 0, stream>>>(bsum_dis, bl_l + 0 * DD, bl_l + 1 * DD, bl_l + 3 * DD);

        // 3) scatter-add per relation (src = ei[0], dst = ei[1])
        scatter_add_kernel<<<sgrid, 256, 0, stream>>>(xc, ei_cause,  ei_cause + E,  agg_cause,  cnt_cause,  E);
        scatter_add_kernel<<<sgrid, 256, 0, stream>>>(xc, ei_relate, ei_relate + E, agg_relate, cnt_relate, E);
        scatter_add_kernel<<<sgrid, 256, 0, stream>>>(xd, ei_child,  ei_child + E,  agg_child,  cnt_child,  E);
        scatter_add_kernel<<<sgrid, 256, 0, stream>>>(xd, ei_rev,    ei_rev + E,    agg_rev,    cnt_rev,    E);

        // 4) fused WMMA GEMMs
        const int gdis  = ((Nd >> 4) * 16 + 7) / 8;
        const int gchem = ((Nc >> 4) * 16 + 7) / 8;
        gemm_wmma_kernel<<<gdis, 256, 0, stream>>>(
            agg_cause,  cnt_cause,  pk[0],
            agg_relate, cnt_relate, pk[1],
            agg_child,  cnt_child,  pk[2],
            xd,         nullptr,    pk[4],
            bsum_dis, od, Nd, relu);
        gemm_wmma_kernel<<<gchem, 256, 0, stream>>>(
            agg_rev, cnt_rev, pk[3],
            xc,      nullptr, pk[5],
            nullptr, nullptr, nullptr,
            nullptr, nullptr, nullptr,
            bl_l + 2 * DD, oc, Nc, relu);
    }
}
